// MACEBlock_7275674599652
// MI455X (gfx1250) — matrix operand order repaired and checked by reading, not compiled, not run on hardware
//
#include <hip/hip_runtime.h>
#include <hip/hip_bf16.h>
#include <math.h>

#define N_ATOMS   4096
#define N_EDGES   131072
#define D_MSG     128
#define AGG_COLS  64
#define KT_TP     36      // 1152 / 32 K-tiles for the tensor-product GEMM
#define EPB       64      // edges per block (4 M-tiles, one per wave)

typedef __attribute__((ext_vector_type(16))) _Float16 v16h;
typedef __attribute__((ext_vector_type(2)))  _Float16 v2h;
typedef __attribute__((ext_vector_type(8)))  float    v8f;

__device__ __forceinline__ v8f wmma16x16x32(v16h a, v16h b, v8f c) {
  // (neg_a, A, neg_b, B, c_mod, C, reuse_a, reuse_b)
  return __builtin_amdgcn_wmma_f32_16x16x32_f16(false, a, false, b, (short)0, c,
                                                false, false);
}

// 16-bit A-matrix 16x32 fragment K mapping (contiguous, matches B fragments):
// VGPR i: K = 2i + 16*khalf
__device__ __forceinline__ int amap_base(int khalf, int i) {
  return 2 * i + 16 * khalf;
}

__device__ __forceinline__ float silu_f(float x)    { return x / (1.f + __expf(-x)); }
__device__ __forceinline__ float sigmoid_f(float x) { return 1.f / (1.f + __expf(-x)); }

__device__ __forceinline__ float half_red_max(float v) {
  v = fmaxf(v, __shfl_xor(v, 1));
  v = fmaxf(v, __shfl_xor(v, 2));
  v = fmaxf(v, __shfl_xor(v, 4));
  v = fmaxf(v, __shfl_xor(v, 8));
  return v;
}
__device__ __forceinline__ float half_red_sum(float v) {
  v += __shfl_xor(v, 1);
  v += __shfl_xor(v, 2);
  v += __shfl_xor(v, 4);
  v += __shfl_xor(v, 8);
  return v;
}

// ---------------------------------------------------------------------------
// Weight pre-swizzle: W_tp (permuted to k' = c*9+s ordering) into B-fragment
// layout: out[((kt*4 + nt)*32 + lane)*16 + j], B map K = kt*32 + 16*khalf + j.
// Only first 64 output columns (4 n-tiles) are ever consumed downstream.
// ---------------------------------------------------------------------------
__global__ void swz_wtp(const float* __restrict__ Wtp, _Float16* __restrict__ out) {
  int idx = blockIdx.x * blockDim.x + threadIdx.x;
  if (idx >= KT_TP * 4 * 512) return;
  int j    = idx & 15;
  int lane = (idx >> 4) & 31;
  int tnt  = idx >> 9;
  int kt = tnt >> 2, nt = tnt & 3;
  int kperm = kt * 32 + 16 * (lane >> 4) + j;     // permuted K index: c*9 + s
  int c = kperm / 9, s = kperm - c * 9;
  int orig = (s == 0) ? c
           : ((s < 4) ? (128 + c * 3 + (s - 1))
                      : (512 + c * 5 + (s - 4)));
  out[idx] = (_Float16)Wtp[orig * 128 + nt * 16 + (lane & 15)];
}

// Generic node-weight swizzle (K = 128): [128 x N] f32 -> B-fragment f16
__global__ void swz_node(const float* __restrict__ W, _Float16* __restrict__ out, int N) {
  int ntiles = N >> 4;
  int total  = ntiles << 11;   // 4 ktiles * ntiles * 512 halves
  int idx = blockIdx.x * blockDim.x + threadIdx.x;
  if (idx >= total) return;
  int j    = idx & 15;
  int lane = (idx >> 4) & 31;
  int tnt  = idx >> 9;
  int kt = tnt / ntiles, nt = tnt % ntiles;
  int K = kt * 32 + 16 * (lane >> 4) + j;
  out[idx] = (_Float16)W[K * N + nt * 16 + (lane & 15)];
}

// ---------------------------------------------------------------------------
// Edge kernel: 64 edges per block, 4 waves (one 16-edge M-tile each).
// Featurize + radial MLP (VALU), then K-streamed WMMA GEMM: per 32-wide
// K-tile, stage the 4 KB B chunk (all 4 n-tiles) into double-buffered LDS
// shared by all waves, build the 4 A fragments in LDS, and run 4 WMMAs per
// wave.  B traffic drops 4x vs a 16-edge block.  Epilogue scatter-adds into
// agg[dst][0:64] with global f32 atomics.
// ---------------------------------------------------------------------------
__global__ __launch_bounds__(128)
void edge_kernel(const float* __restrict__ evec, const float* __restrict__ elen,
                 const int* __restrict__ eidx,
                 const float* __restrict__ Wr1, const float* __restrict__ br1,
                 const float* __restrict__ Wr2, const float* __restrict__ br2,
                 const _Float16* __restrict__ WtpS, const float* __restrict__ btp,
                 float* __restrict__ agg) {
  __shared__ float sh_s[EPB][9];
  __shared__ float rbf_s[EPB][8];
  __shared__ float rad_s[EPB][128];
  __shared__ int   dst_s[EPB];
  // h1 scratch (64x64 f32 = 16 KB) is dead once rad_s is built; the same
  // region is then reused as double-buffered A/B fragment staging (8 KB + 8 KB)
  __shared__ __align__(32) unsigned char staging[16384];
  float (*h1_s)[64] = (float (*)[64])staging;
  _Float16* Abuf = (_Float16*)staging;            // [2][4 mtiles][512 halves]
  _Float16* Bbuf = (_Float16*)(staging + 8192);   // [2][4 ntiles][512 halves]

  const int t  = threadIdx.x;
  const int e0 = blockIdx.x * EPB;

  if (t < EPB) {
    int e = e0 + t;
    float vx = evec[e * 3 + 0], vy = evec[e * 3 + 1], vz = evec[e * 3 + 2];
    float r = sqrtf(vx * vx + vy * vy + vz * vz) + 1e-8f;
    float x = vx / r, y = vy / r, z = vz / r;
    sh_s[t][0] = 1.f;  sh_s[t][1] = y;      sh_s[t][2] = z;
    sh_s[t][3] = x;    sh_s[t][4] = 3.f * z * z - 1.f;
    sh_s[t][5] = x * z; sh_s[t][6] = y * z; sh_s[t][7] = x * y;
    sh_s[t][8] = x * x - y * y;
    float len = elen[e];
    const float pic = 0.5235987755982988f;   // pi / 6
    float env  = (len < 6.0f) ? 0.5f * (__cosf(len * pic) + 1.f) : 0.f;
    float invl = env / len;
#pragma unroll
    for (int k = 0; k < 8; ++k)
      rbf_s[t][k] = __sinf(len * pic * (float)(k + 1)) * invl;
    dst_s[t] = eidx[N_EDGES + e];
  }
  __syncthreads();

  { // radial layer 1: [64,8] @ [8,64] + silu ; each thread 32 outputs
    int e = t >> 1, c0 = (t & 1) * 32;
    for (int cc = 0; cc < 32; ++cc) {
      int c = c0 + cc;
      float acc = br1[c];
#pragma unroll
      for (int k = 0; k < 8; ++k) acc += rbf_s[e][k] * Wr1[k * 64 + c];
      h1_s[e][c] = silu_f(acc);
    }
  }
  __syncthreads();

  { // radial layer 2: [64,64] @ [64,128] + silu ; each thread 64 outputs
    int e = t >> 1, c0 = (t & 1) * 64;
    for (int cc = 0; cc < 64; ++cc) {
      int c = c0 + cc;
      float acc = br2[c];
      for (int k = 0; k < 64; ++k) acc += h1_s[e][k] * Wr2[k * 128 + c];
      rad_s[e][c] = silu_f(acc);
    }
  }
  __syncthreads();   // rad_s complete; h1 region now free for A/B staging

  const int w = t >> 5, lane = t & 31;      // consumer role: wave = M-tile
  const int mt   = t >> 5;                  // staging role: A fragment M-tile
  const int lp   = t & 31;                  // staging role: fragment lane
  const int aedge = mt * 16 + (lp & 15);
  const int akh   = lp >> 4;

  // staging helpers (each thread handles exactly one 16-half fragment slice)
  auto stageB = [&](int kt, int bsel) {
    v16h bv = *(const v16h*)(WtpS + (size_t)kt * 2048 + t * 16);
    *(v16h*)(Bbuf + bsel * 2048 + t * 16) = bv;
  };
  auto buildA = [&](int kt, int bsel) {
    v16h av;
#pragma unroll
    for (int j = 0; j < 16; ++j) {
      int K = kt * 32 + amap_base(akh, j >> 1) + (j & 1);   // k' = c*9 + s
      int c = K / 9, s = K - c * 9;
      av[j] = (_Float16)(rad_s[aedge][c] * sh_s[aedge][s]);
    }
    *(v16h*)(Abuf + bsel * 2048 + (mt * 32 + lp) * 16) = av;
  };

  v8f acc[4] = {{}, {}, {}, {}};

  stageB(0, 0);
  buildA(0, 0);
  for (int kt = 0; kt < KT_TP; ++kt) {
    int bsel = kt & 1;
    __syncthreads();                         // buffer bsel fully staged
    if (kt + 1 < KT_TP) {
      stageB(kt + 1, bsel ^ 1);
      buildA(kt + 1, bsel ^ 1);
    }
    if (kt + 2 < KT_TP)                      // global_prefetch_b8 of k+2 chunk
      __builtin_prefetch(WtpS + (size_t)(kt + 2) * 2048, 0, 3);
    v16h a = *(const v16h*)(Abuf + bsel * 2048 + (w * 32 + lane) * 16);
#pragma unroll
    for (int nt = 0; nt < 4; ++nt) {
      v16h b = *(const v16h*)(Bbuf + bsel * 2048 + (nt * 32 + lane) * 16);
      acc[nt] = wmma16x16x32(a, b, acc[nt]);
    }
  }

  // scatter epilogue: wave w owns edges [w*16, w*16+16)
#pragma unroll
  for (int nt = 0; nt < 4; ++nt) {
    int n = nt * 16 + (lane & 15);
    float bv = btp[n];
#pragma unroll
    for (int r = 0; r < 8; ++r) {
      int erow = w * 16 + r + 8 * (lane >> 4);
      atomicAdd(&agg[dst_s[erow] * AGG_COLS + n], acc[nt][r] + bv);
    }
  }
}

// combined = concat(embed[z], agg[:, :64]) -> f16
__global__ void build_combined(const int* __restrict__ an, const float* __restrict__ embed,
                               const float* __restrict__ agg, _Float16* __restrict__ out) {
  int idx = blockIdx.x * blockDim.x + threadIdx.x;   // < 4096 * 128
  int node = idx >> 7, col = idx & 127;
  float v = (col < 64) ? embed[an[node] * 64 + col]
                       : agg[node * AGG_COLS + (col - 64)];
  out[idx] = (_Float16)v;
}

// ---------------------------------------------------------------------------
// Generic node GEMM: [4096 x 128] f16 @ swizzled [128 x N] f16 + bias.
// mode 0: y = v ; mode 1: y = silu(v) ; mode 2: g = sigmoid(v),
//   y = g*aux1 + (1-g)*aux2  (gate mix).
// ---------------------------------------------------------------------------
__global__ __launch_bounds__(128)
void gemm_node(const _Float16* __restrict__ A, const _Float16* __restrict__ Wswz,
               const float* __restrict__ bias, int N, int ntiles, int mode,
               float* __restrict__ outF, _Float16* __restrict__ outH,
               const float* __restrict__ aux1, const float* __restrict__ aux2) {
  const int t = threadIdx.x, w = t >> 5, lane = t & 31;
  const int m = lane & 15, khalf = lane >> 4;
  const int m0 = blockIdx.x * 16;

  v16h af[4];
#pragma unroll
  for (int kt = 0; kt < 4; ++kt) {
#pragma unroll
    for (int i = 0; i < 8; ++i) {
      int kb = kt * 32 + amap_base(khalf, i);
      v2h p = *(const v2h*)(A + (m0 + m) * 128 + kb);
      af[kt][2 * i] = p[0]; af[kt][2 * i + 1] = p[1];
    }
  }

  for (int nt = w; nt < ntiles; nt += 4) {
    v8f acc = {};
#pragma unroll
    for (int kt = 0; kt < 4; ++kt) {
      v16h b = *(const v16h*)(Wswz + (((kt * ntiles + nt) * 32 + lane) << 4));
      acc = wmma16x16x32(af[kt], b, acc);
    }
    int col = nt * 16 + m;
    float bv = bias[col];
#pragma unroll
    for (int r = 0; r < 8; ++r) {
      int row = m0 + r + 8 * khalf;
      float v = acc[r] + bv;
      float y;
      if (mode == 1)      y = silu_f(v);
      else if (mode == 2) {
        float g = sigmoid_f(v);
        y = g * aux1[row * N + col] + (1.f - g) * aux2[row * N + col];
      } else              y = v;
      if (outF) outF[row * N + col] = y;
      if (outH) outH[row * N + col] = (_Float16)y;
    }
  }
}

// V transpose for attention B fragments: vt[h*32+d][key] = qkv[key][256 + h*32 + d]
__global__ void vtrans(const _Float16* __restrict__ qkv, _Float16* __restrict__ vt) {
  int idx = blockIdx.x * blockDim.x + threadIdx.x;   // < 128 * 4096
  int dall = idx >> 12, key = idx & 4095;
  vt[idx] = qkv[key * 384 + 256 + dall];
}

// ---------------------------------------------------------------------------
// Flash attention: one wave = one (head, 16-query tile).  Online softmax with
// half-wave shuffles; P bounced through per-wave LDS (s_wait_dscnt fences).
// ---------------------------------------------------------------------------
__global__ __launch_bounds__(128)
void attn_kernel(const _Float16* __restrict__ qkv, const _Float16* __restrict__ vt,
                 _Float16* __restrict__ outH) {
  __shared__ _Float16 pl[4][512];   // [wave][16 rows x 32 keys]
  const int t = threadIdx.x, w = t >> 5, lane = t & 31;
  const int m = lane & 15, khalf = lane >> 4;
  const int head = blockIdx.y;
  const int q0 = (blockIdx.x * 4 + w) * 16;
  const float scale = 0.17677669529663687f;   // 1/sqrt(32)

  v16h qf;
#pragma unroll
  for (int i = 0; i < 8; ++i) {
    int d = amap_base(khalf, i);
    v2h p = *(const v2h*)(qkv + (q0 + m) * 384 + head * 32 + d);
    qf[2 * i] = p[0]; qf[2 * i + 1] = p[1];
  }

  float mr[8], lr[8];
#pragma unroll
  for (int r = 0; r < 8; ++r) { mr[r] = -1e30f; lr[r] = 0.f; }
  v8f o0 = {}, o1 = {};

  for (int kb = 0; kb < N_ATOMS; kb += 32) {
    // K^T B fragments: lane column = key, contiguous head-dims per lane
    v16h b0 = *(const v16h*)(qkv + (kb + m) * 384 + 128 + head * 32 + 16 * khalf);
    v16h b1 = *(const v16h*)(qkv + (kb + 16 + m) * 384 + 128 + head * 32 + 16 * khalf);
    v8f zc = {};
    v8f s0 = wmma16x16x32(qf, b0, zc);
    v8f s1 = wmma16x16x32(qf, b1, zc);

#pragma unroll
    for (int r = 0; r < 8; ++r) {
      float a = s0[r] * scale, b = s1[r] * scale;
      float cm   = half_red_max(fmaxf(a, b));
      float mnew = fmaxf(mr[r], cm);
      float alpha = __expf(mr[r] - mnew);
      float p0 = __expf(a - mnew);
      float p1 = __expf(b - mnew);
      float rs = half_red_sum(p0 + p1);
      lr[r] = lr[r] * alpha + rs;
      mr[r] = mnew;
      o0[r] *= alpha; o1[r] *= alpha;
      int row = r + 8 * khalf;
      pl[w][row * 32 + m]      = (_Float16)p0;
      pl[w][row * 32 + 16 + m] = (_Float16)p1;
    }
    asm volatile("s_wait_dscnt 0" ::: "memory");   // wave-local store->load fence

    v16h pf;
#pragma unroll
    for (int i = 0; i < 8; ++i) {
      int kk = amap_base(khalf, i);
      v2h p = *(const v2h*)(&pl[w][m * 32 + kk]);
      pf[2 * i] = p[0]; pf[2 * i + 1] = p[1];
    }
    asm volatile("s_wait_dscnt 0" ::: "memory");   // loads done before next stores

    v16h v0 = *(const v16h*)(vt + (head * 32 + m) * 4096 + kb + 16 * khalf);
    v16h v1 = *(const v16h*)(vt + (head * 32 + 16 + m) * 4096 + kb + 16 * khalf);
    o0 = wmma16x16x32(pf, v0, o0);
    o1 = wmma16x16x32(pf, v1, o1);
  }

#pragma unroll
  for (int r = 0; r < 8; ++r) {
    float inv = 1.f / lr[r];
    int row = q0 + r + 8 * khalf;
    outH[row * 128 + head * 32 + m]      = (_Float16)(o0[r] * inv);
    outH[row * 128 + head * 32 + 16 + m] = (_Float16)(o1[r] * inv);
  }
}

// ---------------------------------------------------------------------------
extern "C" void kernel_launch(void* const* d_in, const int* in_sizes, int n_in,
                              void* d_out, int out_size, void* d_ws, size_t ws_size,
                              hipStream_t stream) {
  const int*   atomic_numbers = (const int*)  d_in[0];
  const int*   edge_index     = (const int*)  d_in[2];
  const float* edge_vectors   = (const float*)d_in[3];
  const float* edge_lengths   = (const float*)d_in[4];
  const float* embed_table    = (const float*)d_in[5];
  const float* W_r1 = (const float*)d_in[6];  const float* b_r1 = (const float*)d_in[7];
  const float* W_r2 = (const float*)d_in[8];  const float* b_r2 = (const float*)d_in[9];
  const float* W_tp = (const float*)d_in[10]; const float* b_tp = (const float*)d_in[11];
  const float* W_m1 = (const float*)d_in[12]; const float* b_m1 = (const float*)d_in[13];
  const float* W_m2 = (const float*)d_in[14]; const float* b_m2 = (const float*)d_in[15];
  const float* W_qkv= (const float*)d_in[16]; const float* b_qkv= (const float*)d_in[17];
  const float* W_ao = (const float*)d_in[18]; const float* b_ao = (const float*)d_in[19];
  const float* W_g  = (const float*)d_in[20]; const float* b_g  = (const float*)d_in[21];
  const float* W_o  = (const float*)d_in[22]; const float* b_o  = (const float*)d_in[23];

  uint8_t* p = (uint8_t*)d_ws;
  auto carve = [&](size_t bytes) -> void* {
    void* r = (void*)p;
    p += (bytes + 255) & ~(size_t)255;
    return r;
  };
  float*     agg      = (float*)    carve((size_t)N_ATOMS * AGG_COLS * 4);
  _Float16*  wtpS     = (_Float16*) carve((size_t)KT_TP * 4 * 512 * 2);
  _Float16*  wm1S     = (_Float16*) carve(4 * 8  * 512 * 2);
  _Float16*  wm2S     = (_Float16*) carve(4 * 8  * 512 * 2);
  _Float16*  wqkvS    = (_Float16*) carve(4 * 24 * 512 * 2);
  _Float16*  waoS     = (_Float16*) carve(4 * 8  * 512 * 2);
  _Float16*  wgS      = (_Float16*) carve(4 * 8  * 512 * 2);
  _Float16*  woS      = (_Float16*) carve(4 * 8  * 512 * 2);
  _Float16*  combH    = (_Float16*) carve((size_t)N_ATOMS * 128 * 2);
  _Float16*  h1H      = (_Float16*) carve((size_t)N_ATOMS * 128 * 2);
  _Float16*  updH     = (_Float16*) carve((size_t)N_ATOMS * 128 * 2);
  float*     updF     = (float*)    carve((size_t)N_ATOMS * 128 * 4);
  _Float16*  qkvH     = (_Float16*) carve((size_t)N_ATOMS * 384 * 2);
  _Float16*  vtH      = (_Float16*) carve((size_t)N_ATOMS * 128 * 2);
  _Float16*  attnH    = (_Float16*) carve((size_t)N_ATOMS * 128 * 2);
  float*     attnoF   = (float*)    carve((size_t)N_ATOMS * 128 * 4);
  _Float16*  mixH     = (_Float16*) carve((size_t)N_ATOMS * 128 * 2);

  (void)in_sizes; (void)n_in; (void)out_size; (void)ws_size;

  hipMemsetAsync(agg, 0, (size_t)N_ATOMS * AGG_COLS * 4, stream);

  // weight pre-swizzles (recomputed each call; deterministic)
  swz_wtp<<<(KT_TP * 4 * 512 + 255) / 256, 256, 0, stream>>>(W_tp, wtpS);
  swz_node<<<64,  256, 0, stream>>>(W_m1,  wm1S,  128);
  swz_node<<<64,  256, 0, stream>>>(W_m2,  wm2S,  128);
  swz_node<<<192, 256, 0, stream>>>(W_qkv, wqkvS, 384);
  swz_node<<<64,  256, 0, stream>>>(W_ao,  waoS,  128);
  swz_node<<<64,  256, 0, stream>>>(W_g,   wgS,   128);
  swz_node<<<64,  256, 0, stream>>>(W_o,   woS,   128);

  // edge path: featurize + radial MLP + K-streamed WMMA GEMM + scatter
  edge_kernel<<<N_EDGES / EPB, 128, 0, stream>>>(
      edge_vectors, edge_lengths, edge_index,
      W_r1, b_r1, W_r2, b_r2, wtpS, b_tp, agg);

  build_combined<<<(N_ATOMS * 128) / 256, 256, 0, stream>>>(
      atomic_numbers, embed_table, agg, combH);

  // node MLP chain
  gemm_node<<<N_ATOMS / 16, 128, 0, stream>>>(combH, wm1S, b_m1, 128, 8, 1,
                                              nullptr, h1H, nullptr, nullptr);
  gemm_node<<<N_ATOMS / 16, 128, 0, stream>>>(h1H, wm2S, b_m2, 128, 8, 0,
                                              updF, updH, nullptr, nullptr);
  gemm_node<<<N_ATOMS / 16, 128, 0, stream>>>(updH, wqkvS, b_qkv, 384, 24, 0,
                                              nullptr, qkvH, nullptr, nullptr);

  // attention
  vtrans<<<(N_ATOMS * 128) / 256, 256, 0, stream>>>(qkvH, vtH);
  attn_kernel<<<dim3(N_ATOMS / 64, 4), 128, 0, stream>>>(qkvH, vtH, attnH);

  // output projection + gated mix + final linear
  gemm_node<<<N_ATOMS / 16, 128, 0, stream>>>(attnH, waoS, b_ao, 128, 8, 0,
                                              attnoF, nullptr, nullptr, nullptr);
  gemm_node<<<N_ATOMS / 16, 128, 0, stream>>>(updH, wgS, b_g, 128, 8, 2,
                                              nullptr, mixH, attnoF, updF);
  gemm_node<<<N_ATOMS / 16, 128, 0, stream>>>(mixH, woS, b_o, 128, 8, 0,
                                              (float*)d_out, nullptr, nullptr, nullptr);
}
